// MultiHeadSelfAttention_67345087201751
// MI455X (gfx1250) — matrix operand order repaired and checked by reading, not compiled, not run on hardware
//
#include <hip/hip_runtime.h>
#include <math.h>

// ---------------------------------------------------------------------------
// MHSA for MI455X (gfx1250, wave32): all matmuls via v_wmma_f32_16x16x32_f16.
// ---------------------------------------------------------------------------

typedef __attribute__((ext_vector_type(16))) _Float16 v16h;
typedef __attribute__((ext_vector_type(4)))  _Float16 v4h;
typedef __attribute__((ext_vector_type(8)))  float    v8f;

#define D_MODEL   1024
#define NUM_HEADS 16
#define HEAD_DIM  64
#define BATCH     2
#define SEQ       2048
#define MROWS     (BATCH * SEQ)     // 4096
#define NEG_INF   (-1000000.0f)

union Frag16 { v16h v; unsigned int u[8]; };

static __device__ __forceinline__ v8f wmma_f16(v16h a, v16h b, v8f c) {
    // (neg_a, A, neg_b, B, c_mod, C, reuse_a, reuse_b)
    return __builtin_amdgcn_wmma_f32_16x16x32_f16(false, a, false, b,
                                                  (short)0, c, false, false);
}

static __device__ __forceinline__ v8f zero8() {
    v8f z;
#pragma unroll
    for (int i = 0; i < 8; ++i) z[i] = 0.0f;
    return z;
}

// pack-convert 4 fp32 -> 4 f16 and store as one 8-byte LDS write
static __device__ __forceinline__ void st4h(_Float16* p, float4 d) {
    v4h h;
    h[0] = (_Float16)d.x; h[1] = (_Float16)d.y;
    h[2] = (_Float16)d.z; h[3] = (_Float16)d.w;
    *(v4h*)p = h;
}

// ---------------------------------------------------------------------------
// Generic GEMM: Y(M x N) = A(M x K) @ W(N x K)^T   (M=4096, N=K=1024)
//   AF16 = 0: A is fp32 (X);  AF16 = 1: A is f16 (attention output)
//   EPI  = 0: write f16, head-split (B*H, S, 64) layout, scaled
//   EPI  = 1: write fp32 row-major + bias (final projection)
// Block: 256 threads = 8 waves; tile 128(M) x 64(N), K-step 64 (16 stages).
// ---------------------------------------------------------------------------
template <int AF16, int EPI>
__global__ __launch_bounds__(256)
void gemm_wmma(const void* __restrict__ Aptr, const float* __restrict__ W,
               const float* __restrict__ bias, void* __restrict__ Out,
               float scale)
{
    constexpr int LDA = 72;   // padded LDS stride (halfs), 64-wide K tiles
    constexpr int LDB = 72;
    __shared__ __align__(16) _Float16 smem[128 * LDA + 64 * LDB];
    _Float16* aS = smem;
    _Float16* bS = smem + 128 * LDA;

    const int tid  = threadIdx.x;
    const int lane = tid & 31;
    const int wave = tid >> 5;        // 0..7
    const int half = lane >> 4;       // lane-half (ISA fragment split)
    const int l16  = lane & 15;
    const int n0   = blockIdx.x * 64;
    const int m0   = blockIdx.y * 128;

    v8f c[4];
#pragma unroll
    for (int nb = 0; nb < 4; ++nb) c[nb] = zero8();

    for (int k0 = 0; k0 < D_MODEL; k0 += 64) {
        __syncthreads();
        // ---- stage A tile (128 x 64) into LDS as f16 ----
        if (AF16) {
            const _Float16* Ag = (const _Float16*)Aptr;
#pragma unroll
            for (int i = 0; i < 4; ++i) {
                int idx = i * 256 + tid;          // 0..1023 uint4s (8 halfs)
                int row = idx >> 3;
                int c8  = (idx & 7) * 8;
                uint4 d = *(const uint4*)&Ag[(size_t)(m0 + row) * D_MODEL + k0 + c8];
                *(uint4*)&aS[row * LDA + c8] = d;
            }
        } else {
            const float* Ag = (const float*)Aptr;
#pragma unroll
            for (int i = 0; i < 8; ++i) {
                int idx = i * 256 + tid;          // 0..2047 float4s
                int row = idx >> 4;
                int kk  = (idx & 15) * 4;
                float4 d = *(const float4*)&Ag[(size_t)(m0 + row) * D_MODEL + k0 + kk];
                st4h(&aS[row * LDA + kk], d);
            }
        }
        // ---- stage W tile (64 x 64) fp32 -> f16 ----
#pragma unroll
        for (int i = 0; i < 4; ++i) {
            int idx = i * 256 + tid;              // 0..1023 float4s
            int row = idx >> 4;
            int kk  = (idx & 15) * 4;
            float4 d = *(const float4*)&W[(size_t)(n0 + row) * D_MODEL + k0 + kk];
            st4h(&bS[row * LDB + kk], d);
        }
        __syncthreads();

        // ---- A fragments (two K-groups of 32; contiguous k = 16*half + s order, same as B) ----
        Frag16 a[2];
        const int arow = wave * 16 + l16;
#pragma unroll
        for (int kg = 0; kg < 2; ++kg)
#pragma unroll
            for (int i = 0; i < 8; ++i) {
                int kk = kg * 32 + (half << 4) + (i << 1);
                a[kg].u[i] = *(const unsigned int*)&aS[arow * LDA + kk];
            }
        // ---- 8 WMMAs: 4 N-tiles x 2 K-groups ----
#pragma unroll
        for (int nb = 0; nb < 4; ++nb) {
            const int brow = nb * 16 + l16;       // N index
#pragma unroll
            for (int kg = 0; kg < 2; ++kg) {
                Frag16 b;
                const int kb = kg * 32 + (half << 4);
#pragma unroll
                for (int i = 0; i < 8; ++i)
                    b.u[i] = *(const unsigned int*)&bS[brow * LDB + kb + (i << 1)];
                c[nb] = wmma_f16(a[kg].v, b.v, c[nb]);
            }
        }
    }

    // ---- epilogue (C layout: row = r + 8*half, col = lane%16) ----
#pragma unroll
    for (int nb = 0; nb < 4; ++nb) {
#pragma unroll
        for (int r = 0; r < 8; ++r) {
            const int row_g = m0 + wave * 16 + r + half * 8;
            const int col_g = n0 + nb * 16 + l16;
            const float val = c[nb][r];
            if (EPI == 0) {
                const int b  = row_g >> 11;       // / SEQ
                const int s  = row_g & (SEQ - 1);
                const int h  = col_g >> 6;
                const int hd = col_g & 63;
                ((_Float16*)Out)[(((size_t)(b * NUM_HEADS + h) * SEQ) + s) * HEAD_DIM + hd]
                    = (_Float16)(val * scale);
            } else {
                ((float*)Out)[(size_t)row_g * D_MODEL + col_g] = val + bias[col_g];
            }
        }
    }
}

// ---------------------------------------------------------------------------
// Flash attention: one block per (batch*head, 128 query rows).
// Key loop in tiles of 64; scores and PV via WMMA; online softmax fp32.
// Row mask (q_row >= valid_len -> all scores = -1e6) matches the reference.
// ---------------------------------------------------------------------------
__global__ __launch_bounds__(256)
void flash_attn(const _Float16* __restrict__ Q, const _Float16* __restrict__ K,
                const _Float16* __restrict__ V, const int* __restrict__ vlens,
                _Float16* __restrict__ O)
{
    constexpr int LDK = 72;   // K tile stride (halfs): 36 dwords, conflict-free
    constexpr int LDV = 66;   // transposed V tile stride: 33 dwords
    constexpr int LDP = 66;   // per-wave P tile stride
    __shared__ __align__(16) _Float16 smem[64 * LDK + 64 * LDV + 8 * 16 * LDP];
    _Float16* kS = smem;
    _Float16* vS = smem + 64 * LDK;
    _Float16* pS = smem + 64 * LDK + 64 * LDV;

    const int tid  = threadIdx.x;
    const int lane = tid & 31;
    const int wave = tid >> 5;
    const int half = lane >> 4;
    const int l16  = lane & 15;
    const int bh   = blockIdx.y;              // 0..31 = b*16 + h
    const int q0   = blockIdx.x * 128;
    const int vlen = vlens[bh];

    const size_t base = (size_t)bh * SEQ * HEAD_DIM;

    // --- persistent Q fragments: 16 rows x 64 hd = 2 A-fragments ---
    Frag16 qa[2];
    {
        const int qrow = q0 + wave * 16 + l16;
        const _Float16* qp = Q + base + (size_t)qrow * HEAD_DIM;
#pragma unroll
        for (int g = 0; g < 2; ++g) {
            const int kb = g * 32 + half * 16;
            uint4 lo = *(const uint4*)&qp[kb];        // K = kb .. kb+7
            uint4 hi = *(const uint4*)&qp[kb + 8];    // K = kb+8 .. kb+15
            qa[g].u[0] = lo.x; qa[g].u[1] = lo.y; qa[g].u[2] = lo.z; qa[g].u[3] = lo.w;
            qa[g].u[4] = hi.x; qa[g].u[5] = hi.y; qa[g].u[6] = hi.z; qa[g].u[7] = hi.w;
        }
    }

    v8f o[4];
#pragma unroll
    for (int nb = 0; nb < 4; ++nb) o[nb] = zero8();
    float mrow[8], lrow[8];
#pragma unroll
    for (int r = 0; r < 8; ++r) { mrow[r] = -3.0e38f; lrow[r] = 0.0f; }

    for (int j0 = 0; j0 < SEQ; j0 += 64) {
        __syncthreads();   // all waves done reading kS/vS from previous tile
        // ---- stage K tile (64 x 64), row-major ----
#pragma unroll
        for (int i = 0; i < 2; ++i) {
            const int idx = i * 256 + tid;        // 0..511 uint4s
            const int jr  = idx >> 3;
            const int c8  = (idx & 7) * 8;
            uint4 d = *(const uint4*)&K[base + (size_t)(j0 + jr) * HEAD_DIM + c8];
            *(uint4*)&kS[jr * LDK + c8] = d;
        }
        // ---- stage V tile transposed: vS[hd][j] ----
#pragma unroll
        for (int i = 0; i < 2; ++i) {
            const int idx = i * 256 + tid;
            const int jr  = idx >> 3;
            const int h0  = (idx & 7) * 8;
            uint4 d = *(const uint4*)&V[base + (size_t)(j0 + jr) * HEAD_DIM + h0];
            const _Float16* hp = (const _Float16*)&d;
#pragma unroll
            for (int ii = 0; ii < 8; ++ii)
                vS[(h0 + ii) * LDV + jr] = hp[ii];
        }
        __syncthreads();

        // ---- scores: S(16 x 64) = Q(16 x 64) * K^T, 8 WMMAs ----
        v8f sfr[4];
#pragma unroll
        for (int cb = 0; cb < 4; ++cb) {
            sfr[cb] = zero8();
            const int jrow = cb * 16 + l16;       // key index (N)
#pragma unroll
            for (int g = 0; g < 2; ++g) {
                Frag16 b;
                const int kb = g * 32 + half * 16;
#pragma unroll
                for (int i = 0; i < 8; ++i)
                    b.u[i] = *(const unsigned int*)&kS[jrow * LDK + kb + (i << 1)];
                sfr[cb] = wmma_f16(qa[g].v, b.v, sfr[cb]);
            }
        }

        // ---- online softmax over the 64-key tile ----
#pragma unroll
        for (int r = 0; r < 8; ++r) {
            const int qrow = q0 + wave * 16 + r + half * 8;
            float s[4];
#pragma unroll
            for (int cb = 0; cb < 4; ++cb) s[cb] = sfr[cb][r];
            if (qrow >= vlen) {                    // reference row mask
#pragma unroll
                for (int cb = 0; cb < 4; ++cb) s[cb] = NEG_INF;
            }
            float mx = fmaxf(fmaxf(s[0], s[1]), fmaxf(s[2], s[3]));
#pragma unroll
            for (int off = 1; off <= 8; off <<= 1)
                mx = fmaxf(mx, __shfl_xor(mx, off, 32));
            const float mnew = fmaxf(mrow[r], mx);
            const float sc   = __expf(mrow[r] - mnew);
            float p[4], sum = 0.0f;
#pragma unroll
            for (int cb = 0; cb < 4; ++cb) { p[cb] = __expf(s[cb] - mnew); sum += p[cb]; }
#pragma unroll
            for (int off = 1; off <= 8; off <<= 1)
                sum += __shfl_xor(sum, off, 32);
            lrow[r] = lrow[r] * sc + sum;
            mrow[r] = mnew;
#pragma unroll
            for (int nb = 0; nb < 4; ++nb) o[nb][r] *= sc;
            // C-layout -> per-wave LDS tile (for A-layout re-read)
            const int prow = r + half * 8;
#pragma unroll
            for (int cb = 0; cb < 4; ++cb)
                pS[wave * 16 * LDP + prow * LDP + cb * 16 + l16] = (_Float16)p[cb];
        }
        // Intra-wave LDS RAW only (per-wave P region; DS ops complete in
        // order within a wave) -> a dscnt wait replaces a block barrier.
        asm volatile("s_wait_dscnt 0" ::: "memory");

        // ---- O += P(16 x 64) * V(64 x 64), 8 WMMAs ----
        Frag16 pa[2];
#pragma unroll
        for (int kg = 0; kg < 2; ++kg)
#pragma unroll
            for (int i = 0; i < 8; ++i) {
                const int kk = kg * 32 + (half << 4) + (i << 1);
                pa[kg].u[i] = *(const unsigned int*)&pS[wave * 16 * LDP + l16 * LDP + kk];
            }
#pragma unroll
        for (int nb = 0; nb < 4; ++nb) {
            const int n = nb * 16 + l16;          // hd index
#pragma unroll
            for (int kg = 0; kg < 2; ++kg) {
                Frag16 b;
                const int kb = kg * 32 + half * 16;
#pragma unroll
                for (int i = 0; i < 8; ++i)
                    b.u[i] = *(const unsigned int*)&vS[n * LDV + kb + (i << 1)];
                o[nb] = wmma_f16(pa[kg].v, b.v, o[nb]);
            }
        }
    }

    // ---- normalize and store merged-head f16 output (B, S, D) ----
    const int b = bh >> 4, h = bh & 15;
#pragma unroll
    for (int r = 0; r < 8; ++r) {
        const float inv = 1.0f / lrow[r];
        const int s = q0 + wave * 16 + r + half * 8;
#pragma unroll
        for (int nb = 0; nb < 4; ++nb) {
            const int d = h * HEAD_DIM + nb * 16 + l16;
            O[((size_t)b * SEQ + s) * D_MODEL + d] = (_Float16)(o[nb][r] * inv);
        }
    }
}

// ---------------------------------------------------------------------------
extern "C" void kernel_launch(void* const* d_in, const int* in_sizes, int n_in,
                              void* d_out, int out_size, void* d_ws, size_t ws_size,
                              hipStream_t stream)
{
    (void)in_sizes; (void)n_in; (void)out_size; (void)ws_size;
    const float* X     = (const float*)d_in[0];
    const float* Wq    = (const float*)d_in[1];
    const float* bq    = (const float*)d_in[2];
    const float* Wk    = (const float*)d_in[3];
    const float* bk    = (const float*)d_in[4];
    const float* Wv    = (const float*)d_in[5];
    const float* bv    = (const float*)d_in[6];
    const float* Wo    = (const float*)d_in[7];
    const float* bo    = (const float*)d_in[8];
    const int*   vlens = (const int*)d_in[9];
    float* out = (float*)d_out;

    const size_t elems = (size_t)MROWS * D_MODEL;   // 4 Mi halfs each
    _Float16* qh = (_Float16*)d_ws;                 // ws usage: 32 MB total
    _Float16* kh = qh + elems;
    _Float16* vh = kh + elems;
    _Float16* oh = vh + elems;

    dim3 gg(D_MODEL / 64, MROWS / 128);             // (16, 32)
    // Q projection: fold 1/sqrt(HEAD_DIM) = 0.125 into the epilogue.
    gemm_wmma<0, 0><<<gg, 256, 0, stream>>>(X, Wq, bq, qh, 0.125f);
    gemm_wmma<0, 0><<<gg, 256, 0, stream>>>(X, Wk, bk, kh, 1.0f);
    gemm_wmma<0, 0><<<gg, 256, 0, stream>>>(X, Wv, bv, vh, 1.0f);

    dim3 ga(SEQ / 128, BATCH * NUM_HEADS);          // (16, 32)
    flash_attn<<<ga, 256, 0, stream>>>(qh, kh, vh, vlens, oh);

    gemm_wmma<1, 1><<<gg, 256, 0, stream>>>(oh, Wo, bo, out, 1.0f);
}
